// MultiHeadSelfAttention_83803401879580
// MI455X (gfx1250) — matrix operand order repaired and checked by reading, not compiled, not run on hardware
//
#include <hip/hip_runtime.h>
#include <hip/hip_bf16.h>

// ---------------------------------------------------------------------------
// MI455X (gfx1250) multi-LoRA causal attention.
// Compute-bound (~172 GFLOP vs ~150MB traffic) -> everything through
// v_wmma_f32_16x16x32_bf16. LoRA folded into per-batch effective weights.
// B/K/V tiles staged into LDS with GLOBAL_LOAD_ASYNC_TO_LDS_B128 (ASYNCcnt).
// ---------------------------------------------------------------------------

typedef __bf16 bf16;
typedef __attribute__((ext_vector_type(16))) __bf16 v16bf;
typedef __attribute__((ext_vector_type(8)))  float  v8f;

union Frag {            // one WMMA 16-bit operand: 16 bf16 per lane = 2 x b128
    uint4 u[2];
    v16bf v;
};

#define D_MODEL 2048
#define SEQ     2048
#define BATCH   2
#define NHEAD   16
#define DKV     128
#define RANK    16
#define ROWS    (BATCH * SEQ)   // 4096 flattened (b,s) rows

__device__ __forceinline__ v8f wmma_bf16(const Frag& a, const Frag& b, v8f c) {
    return __builtin_amdgcn_wmma_f32_16x16x32_bf16(
        /*neg_a=*/false, a.v, /*neg_b=*/false, b.v,
        /*c_mod=*/(short)0, c, /*reuse_a=*/false, /*reuse_b=*/false);
}

// Async memory->LDS b128 copy (per enabled lane), tracked with ASYNCcnt.
// LDS operand is the wave-relative LDS byte offset (low 32 bits of the
// generic LDS address).
__device__ __forceinline__ void async_ld_b128(const bf16* gsrc, void* ldst) {
    unsigned l = (unsigned)(uintptr_t)ldst;
    unsigned long long g = (unsigned long long)(uintptr_t)gsrc;
    asm volatile("global_load_async_to_lds_b128 %0, %1, off"
                 :: "v"(l), "v"(g) : "memory");
}
__device__ __forceinline__ void wait_async0() {
    asm volatile("s_wait_asynccnt 0" ::: "memory");
}

// ---------------------------------------------------------------------------
// K1: W_effT[b][d][o] = W[o][d] + sum_r B[id_b][o][r] * A[id_b][r][d]  (bf16)
// Stored K-major ([d][o]) so B tiles (row = K, N contiguous)
// load as contiguous 16B chunks.
// ---------------------------------------------------------------------------
__global__ __launch_bounds__(256) void build_weff(
    const float* __restrict__ W, const float* __restrict__ A,
    const float* __restrict__ B, const int* __restrict__ starts,
    bf16* __restrict__ WT)
{
    const int b   = blockIdx.y;
    const int idx = blockIdx.x * 256 + threadIdx.x;   // d-major element index
    const int o   = idx & (D_MODEL - 1);
    const int d   = idx >> 11;

    // adapter id: searchsorted(starts, b, 'right') - 1, clipped
    int pos = 0;
    #pragma unroll
    for (int r = 0; r < 4; ++r) pos += (starts[r] <= b) ? 1 : 0;
    int id = pos - 1;
    id = id < 0 ? 0 : (id > 3 ? 3 : id);

    float acc = W[(size_t)o * D_MODEL + d];
    const float* Br = B + ((size_t)id * D_MODEL + o) * RANK;
    const float* Ac = A + (size_t)id * RANK * D_MODEL + d;
    #pragma unroll
    for (int r = 0; r < RANK; ++r)
        acc += Br[r] * Ac[(size_t)r * D_MODEL];

    WT[((size_t)b * D_MODEL + d) * D_MODEL + o] = (bf16)acc;
}

// ---------------------------------------------------------------------------
// K2: f32 -> bf16
// ---------------------------------------------------------------------------
__global__ __launch_bounds__(256) void f32_to_bf16(
    const float* __restrict__ in, bf16* __restrict__ out)
{
    const size_t i = (size_t)blockIdx.x * 256 + threadIdx.x;
    out[i] = (bf16)in[i];
}

// ---------------------------------------------------------------------------
// K3/K6: GEMM  Y[row][o] = sum_d X[row][d] * W_effT[batch][d][o]
// Block: 256 threads = 8 waves; macro tile 128 rows x 64 cols.
// Per wave: 16 rows x 4 column-tiles -> 4 v8f accumulators; 16 wmma per
// 128-deep K chunk against 8 global b128 (A) + async-staged LDS B tile.
// Bt rows padded to 72 halfwords: 144B stride (16B aligned, conflict-free
// within a 16-lane half).
// ---------------------------------------------------------------------------
__global__ __launch_bounds__(256) void gemm_bf16(
    const bf16* __restrict__ X,    // [ROWS][D_MODEL] bf16
    const bf16* __restrict__ WT,   // [BATCH][D_MODEL d][D_MODEL o] bf16
    float* __restrict__ Yf, bf16* __restrict__ Yb, int write_f32)
{
    __shared__ bf16 Bt[128][72];   // 18 KB

    const int t    = threadIdx.x;
    const int lane = t & 31, wave = t >> 5;
    const int ln   = lane & 15, hi = lane >> 4;
    const int klo  = hi << 3;
    const int rb   = blockIdx.x;            // 0..31 (row blocks of 128)
    const int ob   = blockIdx.y;            // 0..31 (col blocks of 64)
    const int batch = rb >> 4;              // 16 row-blocks per batch
    const int o0   = ob << 6;
    const int arow = (rb << 7) + (wave << 4) + ln;

    const bf16* Wp   = WT + (size_t)batch * D_MODEL * D_MODEL;
    const bf16* Arow = X + (size_t)arow * D_MODEL;

    v8f acc[4];
    #pragma unroll
    for (int nt = 0; nt < 4; ++nt) acc[nt] = (v8f){0.f,0.f,0.f,0.f,0.f,0.f,0.f,0.f};

    for (int kc = 0; kc < 16; ++kc) {       // 16 K-chunks of 128
        __syncthreads();                    // previous chunk fully consumed
        // async stage: Bt[dd][0..63] = W_effT[kc*128+dd][o0..o0+63]
        // 128 rows x 8 b128 = 1024 copies, 4 per thread
        #pragma unroll
        for (int i = 0; i < 4; ++i) {
            const int lin  = i * 256 + t;
            const int dd   = lin >> 3;
            const int part = (lin & 7) << 3;    // halfword offset in row
            async_ld_b128(Wp + (size_t)(kc * 128 + dd) * D_MODEL + o0 + part,
                          &Bt[dd][part]);
        }
        wait_async0();
        __syncthreads();

        if (kc < 15) __builtin_prefetch(Arow + (kc + 1) * 128, 0, 1);

        #pragma unroll
        for (int k = 0; k < 128; k += 32) {
            Frag a;
            const bf16* ap = Arow + kc * 128 + k + klo;
            a.u[0] = *(const uint4*)(ap);
            a.u[1] = *(const uint4*)(ap + 16);
            const int kk = k + klo;                     // B lane = N column, elements = K
            #pragma unroll
            for (int nt = 0; nt < 4; ++nt) {
                Frag b;
                #pragma unroll
                for (int j = 0; j < 8; ++j) {
                    b.v[j]     = Bt[kk + j][nt * 16 + ln];
                    b.v[8 + j] = Bt[kk + 16 + j][nt * 16 + ln];
                }
                acc[nt] = wmma_bf16(a, b, acc[nt]);
            }
        }
    }

    // C/D layout: VGPR i -> (M = i + 8*hi, N = ln)
    #pragma unroll
    for (int nt = 0; nt < 4; ++nt) {
        #pragma unroll
        for (int i = 0; i < 8; ++i) {
            const int r = (rb << 7) + (wave << 4) + (hi << 3) + i;
            const size_t idx = (size_t)r * D_MODEL + o0 + (nt << 4) + ln;
            if (write_f32) Yf[idx] = acc[nt][i];
            else           Yb[idx] = (bf16)acc[nt][i];
        }
    }
}

// ---------------------------------------------------------------------------
// K4: RoPE (interleaved pairs) + head split.
//   Qh [bh][s][d]   (A-major for scores)
//   Kt [bh][d][s]   (K-major for scores B fragments)
//   Vh [bh][s][d]   (K-major for PV B fragments)
// ---------------------------------------------------------------------------
__global__ __launch_bounds__(64) void rope_split(
    const bf16* __restrict__ Yq, const bf16* __restrict__ Yk,
    const bf16* __restrict__ Yv,
    bf16* __restrict__ Qh, bf16* __restrict__ Kt, bf16* __restrict__ Vh)
{
    const int s = blockIdx.x, h = blockIdx.y, b = blockIdx.z;
    const int i = threadIdx.x;                       // pair index 0..63
    const size_t src = ((size_t)(b * SEQ + s)) * D_MODEL + h * DKV;

    const float q1 = (float)Yq[src + 2*i], q2 = (float)Yq[src + 2*i + 1];
    const float k1 = (float)Yk[src + 2*i], k2 = (float)Yk[src + 2*i + 1];
    const float v1 = (float)Yv[src + 2*i], v2 = (float)Yv[src + 2*i + 1];

    const float freq = __powf(10000.0f, -(float)(2 * i) * (1.0f / 128.0f));
    const float ang  = (float)s * freq;
    float sn, cs;
    __sincosf(ang, &sn, &cs);

    const int bh = b * NHEAD + h;
    const size_t qdst = ((size_t)bh * SEQ + s) * DKV;
    Qh[qdst + 2*i]     = (bf16)(q1 * cs - q2 * sn);
    Qh[qdst + 2*i + 1] = (bf16)(q1 * sn + q2 * cs);
    Vh[qdst + 2*i]     = (bf16)v1;
    Vh[qdst + 2*i + 1] = (bf16)v2;

    const size_t kdst = (size_t)bh * DKV * SEQ;
    Kt[kdst + (size_t)(2*i)     * SEQ + s] = (bf16)(k1 * cs - k2 * sn);
    Kt[kdst + (size_t)(2*i + 1) * SEQ + s] = (bf16)(k1 * sn + k2 * cs);
}

// ---------------------------------------------------------------------------
// K5: causal flash attention. Block = 8 waves = 128 q rows; per 32-key block
// the whole workgroup shares async-staged LDS tiles of K^T (128x32) and
// V (32x128). Per wave: 8 WMMA for S = Q*K^T, online softmax (shfl-xor row
// reductions within 16-lane halves), P staged via LDS (C-layout -> A-layout),
// 8 WMMA for O += P*V. Waves past their causal extent skip compute but join
// barriers. ctx written bf16 in [b][s][h*128+d] for the o-projection.
// ---------------------------------------------------------------------------
__global__ __launch_bounds__(256) void attn_kernel(
    const bf16* __restrict__ Qh, const bf16* __restrict__ Kt,
    const bf16* __restrict__ Vh, bf16* __restrict__ ctx)
{
    __shared__ bf16 Ktile[128][56];    // 14.0 KB (112B stride, 16B aligned)
    __shared__ bf16 Vtile[32][136];    //  8.5 KB (272B stride, 16B aligned)
    __shared__ bf16 Plds[8][16][56];   // 14.0 KB per-wave P transpose buffers

    const int t = threadIdx.x, lane = t & 31, wave = t >> 5;
    const int ln = lane & 15, hi = lane >> 4, klo = hi << 3;
    const int bh = blockIdx.y, b = bh >> 4, h = bh & 15;
    const int qb0 = blockIdx.x << 7;
    const int q0  = qb0 + (wave << 4);

    const bf16* Qp = Qh + (size_t)bh * SEQ * DKV;
    const bf16* Kp = Kt + (size_t)bh * DKV * SEQ;
    const bf16* Vp = Vh + (size_t)bh * SEQ * DKV;

    // Q fragments cached for the whole k-loop (d = 128 -> 4 K-steps of 32)
    Frag qf[4];
    const bf16* qrow = Qp + (size_t)(q0 + ln) * DKV;
    #pragma unroll
    for (int st = 0; st < 4; ++st) {
        qf[st].u[0] = *(const uint4*)(qrow + st * 32 + klo);
        qf[st].u[1] = *(const uint4*)(qrow + st * 32 + klo + 16);
    }

    v8f o[8];
    #pragma unroll
    for (int nt = 0; nt < 8; ++nt) o[nt] = (v8f){0.f,0.f,0.f,0.f,0.f,0.f,0.f,0.f};
    float mrow[8], lrow[8];
    #pragma unroll
    for (int i = 0; i < 8; ++i) { mrow[i] = -1e30f; lrow[i] = 0.f; }

    const float scale = 0.08838834764831845f;   // 1/sqrt(128)
    const int nkb_wave = (q0 + 47) >> 5;        // this wave's causal extent
    const int nkb_blk  = (qb0 + 159) >> 5;      // block max extent

    for (int kb = 0; kb < nkb_blk; ++kb) {
        const int k0 = kb << 5;

        __syncthreads();                        // tiles fully consumed
        // stage K^T tile: Ktile[d][0..31] = Kt[d][k0..k0+31]; 512 b128
        #pragma unroll
        for (int i = 0; i < 2; ++i) {
            const int lin  = i * 256 + t;
            const int dd   = lin >> 2;
            const int part = (lin & 3) << 3;
            async_ld_b128(Kp + (size_t)dd * SEQ + k0 + part, &Ktile[dd][part]);
        }
        // stage V tile: Vtile[kk][0..127] = Vh[k0+kk][0..127]; 512 b128
        #pragma unroll
        for (int i = 0; i < 2; ++i) {
            const int lin  = i * 256 + t;
            const int kk   = lin >> 4;
            const int part = (lin & 15) << 3;
            async_ld_b128(Vp + (size_t)(k0 + kk) * DKV + part, &Vtile[kk][part]);
        }
        wait_async0();
        __syncthreads();

        if (kb >= nkb_wave) continue;           // beyond causal extent

        v8f s0 = (v8f){0.f,0.f,0.f,0.f,0.f,0.f,0.f,0.f};
        v8f s1 = s0;
        #pragma unroll
        for (int st = 0; st < 4; ++st) {
            Frag b0, b1;
            const int drow = (st << 5) + klo;              // B lane = key column, elements = d
            #pragma unroll
            for (int j = 0; j < 8; ++j) {
                b0.v[j]     = Ktile[drow + j][ln];
                b0.v[8 + j] = Ktile[drow + 16 + j][ln];
                b1.v[j]     = Ktile[drow + j][16 + ln];
                b1.v[8 + j] = Ktile[drow + 16 + j][16 + ln];
            }
            s0 = wmma_bf16(qf[st], b0, s0);
            s1 = wmma_bf16(qf[st], b1, s1);
        }

        float alpha[8];
        #pragma unroll
        for (int i = 0; i < 8; ++i) {
            const int qr = q0 + (hi << 3) + i;             // query row of this VGPR
            float v0 = s0[i] * scale; if (k0 + ln      > qr) v0 = -1e30f;
            float v1 = s1[i] * scale; if (k0 + 16 + ln > qr) v1 = -1e30f;

            float mx = fmaxf(v0, v1);
            #pragma unroll
            for (int off = 1; off < 16; off <<= 1)
                mx = fmaxf(mx, __shfl_xor(mx, off, 16));

            const float mnew = fmaxf(mrow[i], mx);
            const float a    = __expf(mrow[i] - mnew);
            const float p0   = __expf(v0 - mnew);
            const float p1   = __expf(v1 - mnew);
            float ps = p0 + p1;
            #pragma unroll
            for (int off = 1; off < 16; off <<= 1)
                ps += __shfl_xor(ps, off, 16);

            lrow[i]  = lrow[i] * a + ps;
            mrow[i]  = mnew;
            alpha[i] = a;

            Plds[wave][(hi << 3) + i][ln]      = (bf16)p0;
            Plds[wave][(hi << 3) + i][16 + ln] = (bf16)p1;
        }

        #pragma unroll
        for (int nt = 0; nt < 8; ++nt) {
            #pragma unroll
            for (int i = 0; i < 8; ++i) o[nt][i] *= alpha[i];
        }

        // re-read P in A-fragment layout (lane = row ln, K chunks klo/klo+16)
        Frag pa;
        pa.u[0] = *(const uint4*)(&Plds[wave][ln][klo]);
        pa.u[1] = *(const uint4*)(&Plds[wave][ln][klo + 16]);

        const int keyrowL = klo;                           // B lane = d column, elements = key
        #pragma unroll
        for (int nt = 0; nt < 8; ++nt) {
            Frag vb;
            #pragma unroll
            for (int j = 0; j < 8; ++j) {
                vb.v[j]     = Vtile[keyrowL + j][(nt << 4) + ln];
                vb.v[8 + j] = Vtile[keyrowL + 16 + j][(nt << 4) + ln];
            }
            o[nt] = wmma_bf16(pa, vb, o[nt]);
        }
    }

    #pragma unroll
    for (int nt = 0; nt < 8; ++nt) {
        #pragma unroll
        for (int i = 0; i < 8; ++i) {
            const int q   = q0 + (hi << 3) + i;
            const int col = h * DKV + (nt << 4) + ln;
            ctx[((size_t)b * SEQ + q) * D_MODEL + col] = (bf16)(o[nt][i] / lrow[i]);
        }
    }
}

// ---------------------------------------------------------------------------
// Orchestration. Workspace layout: needs ~192 MB.
// ---------------------------------------------------------------------------
extern "C" void kernel_launch(void* const* d_in, const int* in_sizes, int n_in,
                              void* d_out, int out_size, void* d_ws, size_t ws_size,
                              hipStream_t stream)
{
    (void)in_sizes; (void)n_in; (void)out_size; (void)ws_size;

    const float* x      = (const float*)d_in[0];
    const int*   starts = (const int*)  d_in[1];
    const float* Wq = (const float*)d_in[2];
    const float* Aq = (const float*)d_in[3];
    const float* Bq = (const float*)d_in[4];
    const float* Wk = (const float*)d_in[5];
    const float* Ak = (const float*)d_in[6];
    const float* Bk = (const float*)d_in[7];
    const float* Wv = (const float*)d_in[8];
    const float* Av = (const float*)d_in[9];
    const float* Bv = (const float*)d_in[10];
    const float* Wo = (const float*)d_in[11];
    const float* Ao = (const float*)d_in[12];
    const float* Bo = (const float*)d_in[13];
    float* out = (float*)d_out;

    char* ws = (char*)d_ws;
    const size_t MB = 1024 * 1024;
    bf16* WTq = (bf16*)(ws +   0 * MB);   // [2][2048][2048] each = 16MB
    bf16* WTk = (bf16*)(ws +  16 * MB);
    bf16* WTv = (bf16*)(ws +  32 * MB);
    bf16* WTo = (bf16*)(ws +  48 * MB);
    bf16* xb  = (bf16*)(ws +  64 * MB);   // [4096][2048]
    bf16* Yq  = (bf16*)(ws +  80 * MB);   // [4096][2048]
    bf16* Yk  = (bf16*)(ws +  96 * MB);
    bf16* Yv  = (bf16*)(ws + 112 * MB);
    bf16* Qh  = (bf16*)(ws + 128 * MB);   // [32][2048][128]
    bf16* Kt  = (bf16*)(ws + 144 * MB);   // [32][128][2048]
    bf16* Vh  = (bf16*)(ws + 160 * MB);   // [32][2048][128]
    bf16* ctx = (bf16*)(ws + 176 * MB);   // [4096][2048]

    const dim3 weffGrid(D_MODEL * D_MODEL / 256, BATCH);
    build_weff<<<weffGrid, 256, 0, stream>>>(Wq, Aq, Bq, starts, WTq);
    build_weff<<<weffGrid, 256, 0, stream>>>(Wk, Ak, Bk, starts, WTk);
    build_weff<<<weffGrid, 256, 0, stream>>>(Wv, Av, Bv, starts, WTv);
    build_weff<<<weffGrid, 256, 0, stream>>>(Wo, Ao, Bo, starts, WTo);

    f32_to_bf16<<<(ROWS * D_MODEL) / 256, 256, 0, stream>>>(x, xb);

    const dim3 gemmGrid(ROWS / 128, D_MODEL / 64);
    gemm_bf16<<<gemmGrid, 256, 0, stream>>>(xb, WTq, nullptr, Yq, 0);
    gemm_bf16<<<gemmGrid, 256, 0, stream>>>(xb, WTk, nullptr, Yk, 0);
    gemm_bf16<<<gemmGrid, 256, 0, stream>>>(xb, WTv, nullptr, Yv, 0);

    rope_split<<<dim3(SEQ, NHEAD, BATCH), 64, 0, stream>>>(Yq, Yk, Yv, Qh, Kt, Vh);

    attn_kernel<<<dim3(SEQ / 128, BATCH * NHEAD), 256, 0, stream>>>(Qh, Kt, Vh, ctx);

    gemm_bf16<<<gemmGrid, 256, 0, stream>>>(ctx, WTo, out, nullptr, 1);
}
